// PAINN_5987184410901
// MI455X (gfx1250) — matrix operand order repaired and checked by reading, not compiled, not run on hardware
//
#include <hip/hip_runtime.h>
#include <math.h>

#define N_NODES 20000
#define N_EDGES 320000
#define N_GRAPHS 64
#define DDIM    128
#define N_RBF   20
#define RCUT    10.0f
#define PI_F    3.14159265358979323846f

typedef __attribute__((ext_vector_type(16))) __bf16 v16bf;
typedef __attribute__((ext_vector_type(8)))  float  v8f;

union Frag16 { v16bf v; unsigned u[8]; uint4 q[2]; };

__device__ __forceinline__ unsigned pack_bf16(float lo, float hi) {
    unsigned a = __float_as_uint(lo);
    unsigned b = __float_as_uint(hi);
    return (a >> 16) | (b & 0xFFFF0000u);
}
__device__ __forceinline__ unsigned short trunc_bf16(float x) {
    return (unsigned short)(__float_as_uint(x) >> 16);
}

// Build 16x32 bf16 A fragment from LDS staged as packed bf16 pairs (dwords).
// ldRowPairs = row stride in dwords (= row elements / 2), kbPairs = K-chunk base / 2.
// ISA 7.12.2: lanes 0-15 -> row M=lane, K {kb+0..7, kb+16..23};
//             lanes 16-31 -> row M=lane-16, K {kb+8..15, kb+24..31}.
// => two ds_load_b128 per fragment, zero repacking VALU.
__device__ __forceinline__ v16bf build_a_p(const unsigned* ldsPairs, int ldRowPairs,
                                           int kbPairs, int lane) {
    int row = lane & 15;
    const unsigned* p = ldsPairs + row * ldRowPairs + kbPairs + ((lane & 16) ? 4 : 0);
    Frag16 f;
    f.q[0] = *(const uint4*)(p);      // K pairs +0..3   (K kb..kb+7   / kb+8..15)
    f.q[1] = *(const uint4*)(p + 8);  // K pairs +8..11  (K kb+16..23  / kb+24..31)
    return f.v;
}

// Pre-packed B fragments: dword (kc, nt, lane, j) = bf16 pair
// (N = nt*16 + lane%16, K = kc*32 + table-order k of elements 2j, 2j+1). Each lane loads 32 contiguous bytes.
__device__ __forceinline__ v16bf load_b(const unsigned* Bp, int NT, int kc, int nt, int lane) {
    const uint4* p = (const uint4*)(Bp + ((((kc * NT + nt) * 32) + lane) << 3));
    Frag16 f;
    f.q[0] = p[0];
    f.q[1] = p[1];
    return f.v;
}

__device__ __forceinline__ v8f wmma_bf16(v16bf a, v16bf b, v8f c) {
    return __builtin_amdgcn_wmma_f32_16x16x32_bf16(false, a, false, b, (short)0, c, false, false);
}

__device__ __forceinline__ float silu_f(float x) { return x / (1.0f + __expf(-x)); }

// ---------------- weight packing (runs once per launch, tiny) ----------------
__global__ void k_pack_weights(const float* __restrict__ W, unsigned* __restrict__ out,
                               int K, int N) {
    int idx = blockIdx.x * blockDim.x + threadIdx.x;
    int NT = N >> 4;
    int total = (K >> 5) * NT * 256;
    if (idx >= total) return;
    int j = idx & 7;
    int lane = (idx >> 3) & 31;
    int t = idx >> 8;               // kc*NT + nt
    int kc = t / NT, nt = t - kc * NT;
    int k = kc * 32 + ((j < 4) ? (8 * (lane >> 4) + 2 * j) : (16 + 8 * (lane >> 4) + 2 * (j - 4)));
    int n = nt * 16 + (lane & 15);
    out[idx] = pack_bf16(W[(size_t)k * N + n], W[(size_t)(k + 1) * N + n]);
}

// ---------------- init / zero ----------------
__global__ void k_zero(float* __restrict__ p, long n) {
    long i = (long)blockIdx.x * blockDim.x + threadIdx.x;
    if (i < n) p[i] = 0.0f;
}

__global__ void k_init_s(const int* __restrict__ atype, const float* __restrict__ emb,
                         float* __restrict__ s_buf) {
    long i = (long)blockIdx.x * blockDim.x + threadIdx.x;
    if (i >= (long)N_NODES * DDIM) return;
    int node = (int)(i >> 7), c = (int)(i & 127);
    s_buf[i] = emb[atype[node] * DDIM + c];
}

// ---------------- edge message kernel: 1 wave = 16 edges ----------------
// LDS: Wt f32 24576 | sP packed 4096 | hB bf16 4096 | rbf 1280 | dirs 192 | fc 64 | idx 128
#define EDGE_SMEM_BYTES (16*384*4 + 16*64*4 + 16*128*2 + 16*N_RBF*4 + 48*4 + 16*4 + 32*4)
__global__ void __launch_bounds__(32)
k_edge(const int* __restrict__ esrc, const int* __restrict__ edst,
       const float* __restrict__ evec,
       const float* __restrict__ s_buf, const float* __restrict__ v_buf,
       float* __restrict__ ds, float* __restrict__ dv,
       const unsigned* __restrict__ Bp1, const float* __restrict__ phi_b1,
       const unsigned* __restrict__ Bp2, const float* __restrict__ phi_b2,
       const float* __restrict__ filt_w, const float* __restrict__ filt_b) {
    extern __shared__ char smraw[];
    float*          Wt   = (float*)smraw;                    // 16*384 f32
    unsigned*       sP   = (unsigned*)(Wt + 16 * 384);       // 16*64 bf16 pairs
    unsigned short* hB   = (unsigned short*)(sP + 16 * 64);  // 16*128 bf16
    float*          rbf  = (float*)(hB + 16 * DDIM);         // 16*20
    float*          dirs = rbf + 16 * N_RBF;                 // 16*3
    float*          fc   = dirs + 48;                        // 16
    int*            srcI = (int*)(fc + 16);                  // 16
    int*            dstI = srcI + 16;                        // 16

    const int lane = threadIdx.x;
    const int e0 = blockIdx.x * 16;
    const int nl = lane & 15;
    const int mh = (lane & 16) ? 8 : 0;

    if (lane < 16) {
        int e = e0 + lane;
        srcI[lane] = esrc[e];
        dstI[lane] = edst[e];
        float x = evec[3 * e + 0], y = evec[3 * e + 1], z = evec[3 * e + 2];
        float r = sqrtf(x * x + y * y + z * z);
        float inv = 1.0f / r;
        dirs[lane * 3 + 0] = x * inv;
        dirs[lane * 3 + 1] = y * inv;
        dirs[lane * 3 + 2] = z * inv;
        fc[lane] = (r < RCUT) ? 0.5f * (__cosf(PI_F * r / RCUT) + 1.0f) : 0.0f;
#pragma unroll
        for (int k = 0; k < N_RBF; ++k)
            rbf[lane * N_RBF + k] = __sinf((float)(k + 1) * PI_F * r / RCUT) * inv;
    }
    __syncthreads();

    // Recompute filter W = (rbf @ filt_w + b) * fcut for this tile (avoids a 492 MB buffer).
    for (int row = 0; row < 16; ++row) {
        float f = fc[row];
#pragma unroll
        for (int j = 0; j < 12; ++j) {
            int col = j * 32 + lane;
            float acc = filt_b[col];
#pragma unroll
            for (int k = 0; k < N_RBF; ++k)
                acc += rbf[row * N_RBF + k] * filt_w[k * 384 + col];  // rbf: LDS broadcast
            Wt[row * 384 + col] = acc * f;
        }
    }
    // Gather s[src] rows into LDS, packing f32 -> bf16 pairs on the way.
    for (int i = lane; i < 16 * 32; i += 32) {
        int row = i >> 5, c4 = i & 31;
        float4 t = ((const float4*)(s_buf + (size_t)srcI[row] * DDIM))[c4];
        uint2 pp;
        pp.x = pack_bf16(t.x, t.y);
        pp.y = pack_bf16(t.z, t.w);
        ((uint2*)sP)[row * 32 + c4] = pp;
    }
    __syncthreads();

    v16bf a1[4];
#pragma unroll
    for (int kc = 0; kc < 4; ++kc) a1[kc] = build_a_p(sP, 64, kc * 16, lane);

    // layer 1: h = silu(s @ phi_w1 + b1)   (32 WMMAs)
#pragma unroll
    for (int nt = 0; nt < 8; ++nt) {
        v8f acc = {};
#pragma unroll
        for (int kc = 0; kc < 4; ++kc) acc = wmma_bf16(a1[kc], load_b(Bp1, 8, kc, nt, lane), acc);
        float b = phi_b1[nt * 16 + nl];
#pragma unroll
        for (int r = 0; r < 8; ++r)
            hB[(mh + r) * DDIM + nt * 16 + nl] = trunc_bf16(silu_f(acc[r] + b));
    }
    __syncthreads();

    v16bf a2[4];
#pragma unroll
    for (int kc = 0; kc < 4; ++kc) a2[kc] = build_a_p((const unsigned*)hB, 64, kc * 16, lane);

    // layer 2, m1/m3 tiles (64 WMMAs) -> dv scatter
    for (int cnt = 0; cnt < 8; ++cnt) {
        v8f p1 = {}, p3 = {};
#pragma unroll
        for (int kc = 0; kc < 4; ++kc) p1 = wmma_bf16(a2[kc], load_b(Bp2, 24, kc, cnt, lane), p1);
#pragma unroll
        for (int kc = 0; kc < 4; ++kc) p3 = wmma_bf16(a2[kc], load_b(Bp2, 24, kc, 16 + cnt, lane), p3);
        int c = cnt * 16 + nl;
        float b1v = phi_b2[c], b3v = phi_b2[256 + c];
#pragma unroll
        for (int r = 0; r < 8; ++r) {
            int M = mh + r;
            float m1v = (p1[r] + b1v) * Wt[M * 384 + c];
            float m3v = (p3[r] + b3v) * Wt[M * 384 + 256 + c];
            size_t sb = (size_t)srcI[M] * 384;
            size_t db = (size_t)dstI[M] * 384;
#pragma unroll
            for (int d = 0; d < 3; ++d) {
                float val = v_buf[sb + d * DDIM + c] * m1v + dirs[M * 3 + d] * m3v;
                atomicAdd(&dv[db + d * DDIM + c], val);
            }
        }
    }
    // layer 2, m2 tiles (32 WMMAs) -> ds scatter
    for (int cnt = 0; cnt < 8; ++cnt) {
        v8f p2 = {};
#pragma unroll
        for (int kc = 0; kc < 4; ++kc) p2 = wmma_bf16(a2[kc], load_b(Bp2, 24, kc, 8 + cnt, lane), p2);
        int c = cnt * 16 + nl;
        float bv = phi_b2[128 + c];
#pragma unroll
        for (int r = 0; r < 8; ++r) {
            int M = mh + r;
            float m2v = (p2[r] + bv) * Wt[M * 384 + 128 + c];
            atomicAdd(&ds[(size_t)dstI[M] * DDIM + c], m2v);
        }
    }
}

// ---------------- node update kernel: 1 wave = 16 nodes ----------------
// LDS: vB bf16 12288 (reused: hB 4096 + UVs f32 8192) | US f32 24576 | catB bf16 8192
#define NODE_SMEM_BYTES (3*16*128*2 + 3*16*128*4 + 16*256*2)
__global__ void __launch_bounds__(32)
k_node(float* __restrict__ s_buf, float* __restrict__ v_buf,
       const float* __restrict__ ds, const float* __restrict__ dv,
       const unsigned* __restrict__ BpU, const unsigned* __restrict__ BpV,
       const unsigned* __restrict__ BpW1, const float* __restrict__ upd_b1,
       const unsigned* __restrict__ BpW2, const float* __restrict__ upd_b2) {
    extern __shared__ char smraw[];
    unsigned short* vB   = (unsigned short*)smraw;            // 3*16*128 bf16
    float*          US   = (float*)(smraw + 12288);           // 3*16*128 f32
    unsigned short* catB = (unsigned short*)(smraw + 12288 + 24576);  // 16*256 bf16

    const int lane = threadIdx.x;
    const int n0 = blockIdx.x * 16;
    const int nl = lane & 15;
    const int mh = (lane & 16) ? 8 : 0;

    // v_mid = v + dv  (stage bf16 + write back f32); s_mid = s + ds (stage bf16 only)
    for (int d = 0; d < 3; ++d)
        for (int i = lane; i < 16 * DDIM; i += 32) {
            int row = i >> 7, c = i & 127;
            size_t g = (size_t)(n0 + row) * 384 + d * DDIM + c;
            float val = v_buf[g] + dv[g];
            v_buf[g] = val;
            vB[d * 2048 + row * DDIM + c] = trunc_bf16(val);
        }
    for (int i = lane; i < 16 * DDIM; i += 32) {
        int row = i >> 7, c = i & 127;
        size_t g = (size_t)(n0 + row) * DDIM + c;
        catB[row * 256 + 128 + c] = trunc_bf16(s_buf[g] + ds[g]);
    }
    __syncthreads();

    v16bf aV[3][4];
#pragma unroll
    for (int d = 0; d < 3; ++d)
#pragma unroll
        for (int kc = 0; kc < 4; ++kc)
            aV[d][kc] = build_a_p((const unsigned*)(vB + d * 2048), 64, kc * 16, lane);
    __syncthreads();

    unsigned short* hB  = (unsigned short*)smraw;      // reuse vB[0..4096)
    float*          UVs = (float*)(smraw + 4096);      // reuse vB[4096..12288)

    // U = v@U_w, Vp = v@V_w; UV = sum_d U*Vp; Vn = ||Vp||  (192 WMMAs)
#pragma unroll
    for (int nt = 0; nt < 8; ++nt) {
        v8f uv = {}, vn = {};
#pragma unroll
        for (int d = 0; d < 3; ++d) {
            v8f aU = {}, aP = {};
#pragma unroll
            for (int kc = 0; kc < 4; ++kc) {
                aU = wmma_bf16(aV[d][kc], load_b(BpU, 8, kc, nt, lane), aU);
                aP = wmma_bf16(aV[d][kc], load_b(BpV, 8, kc, nt, lane), aP);
            }
            uv += aU * aP;
            vn += aP * aP;
#pragma unroll
            for (int r = 0; r < 8; ++r) US[d * 2048 + (mh + r) * DDIM + nt * 16 + nl] = aU[r];
        }
#pragma unroll
        for (int r = 0; r < 8; ++r) {
            int M = mh + r;
            UVs[M * DDIM + nt * 16 + nl] = uv[r];
            catB[M * 256 + nt * 16 + nl] = trunc_bf16(sqrtf(vn[r]));
        }
    }
    __syncthreads();

    // layer 1: h = silu(cat @ upd_w1 + b1)   (64 WMMAs, K=256)
    v16bf aC[8];
#pragma unroll
    for (int kc = 0; kc < 8; ++kc) aC[kc] = build_a_p((const unsigned*)catB, 128, kc * 16, lane);
#pragma unroll
    for (int nt = 0; nt < 8; ++nt) {
        v8f acc = {};
#pragma unroll
        for (int kc = 0; kc < 8; ++kc) acc = wmma_bf16(aC[kc], load_b(BpW1, 8, kc, nt, lane), acc);
        float b = upd_b1[nt * 16 + nl];
#pragma unroll
        for (int r = 0; r < 8; ++r)
            hB[(mh + r) * DDIM + nt * 16 + nl] = trunc_bf16(silu_f(acc[r] + b));
    }
    __syncthreads();

    v16bf aH[4];
#pragma unroll
    for (int kc = 0; kc < 4; ++kc) aH[kc] = build_a_p((const unsigned*)hB, 64, kc * 16, lane);

    // a_vv tiles -> v += U * avv   (32 WMMAs)
    for (int cnt = 0; cnt < 8; ++cnt) {
        v8f avv = {};
#pragma unroll
        for (int kc = 0; kc < 4; ++kc) avv = wmma_bf16(aH[kc], load_b(BpW2, 24, kc, cnt, lane), avv);
        int c = cnt * 16 + nl;
        float bb = upd_b2[c];
#pragma unroll
        for (int r = 0; r < 8; ++r) {
            int M = mh + r;
            float a = avv[r] + bb;
            size_t g = (size_t)(n0 + M) * 384;
#pragma unroll
            for (int d = 0; d < 3; ++d)
                v_buf[g + d * DDIM + c] += US[d * 2048 + M * DDIM + c] * a;
        }
    }
    // a_sv / a_ss tiles -> s = s_mid + ass + UV*asv  (64 WMMAs)
    // s_mid re-read from global (f32 precision; L2-resident) instead of bf16 LDS copy.
    for (int cnt = 0; cnt < 8; ++cnt) {
        v8f asv = {}, ass = {};
#pragma unroll
        for (int kc = 0; kc < 4; ++kc) asv = wmma_bf16(aH[kc], load_b(BpW2, 24, kc, 8 + cnt, lane), asv);
#pragma unroll
        for (int kc = 0; kc < 4; ++kc) ass = wmma_bf16(aH[kc], load_b(BpW2, 24, kc, 16 + cnt, lane), ass);
        int c = cnt * 16 + nl;
        float bsv = upd_b2[128 + c], bss = upd_b2[256 + c];
#pragma unroll
        for (int r = 0; r < 8; ++r) {
            int M = mh + r;
            size_t g = (size_t)(n0 + M) * DDIM + c;
            float smid = s_buf[g] + ds[g];
            s_buf[g] = smid + (ass[r] + bss) + UVs[M * DDIM + c] * (asv[r] + bsv);
        }
    }
}

// ---------------- readout ----------------
__global__ void k_graph_accum(const float* __restrict__ s_buf, const int* __restrict__ gidx,
                              float* __restrict__ g) {
    long i = (long)blockIdx.x * blockDim.x + threadIdx.x;
    if (i >= (long)N_NODES * DDIM) return;
    int node = (int)(i >> 7), c = (int)(i & 127);
    atomicAdd(&g[gidx[node] * DDIM + c], s_buf[i]);
}

__global__ void k_readout(const float* __restrict__ g,
                          const float* __restrict__ w1, const float* __restrict__ b1,
                          const float* __restrict__ w2, const float* __restrict__ b2,
                          float* __restrict__ out) {
    __shared__ float red[DDIM];
    int gi = blockIdx.x, c = threadIdx.x;
    float acc = b1[c];
    for (int k = 0; k < DDIM; ++k) acc += g[gi * DDIM + k] * w1[k * DDIM + c];
    red[c] = silu_f(acc) * w2[c];
    __syncthreads();
    for (int s = 64; s > 0; s >>= 1) {
        if (c < s) red[c] += red[c + s];
        __syncthreads();
    }
    if (c == 0) out[gi] = red[0] + b2[0];
}

// ---------------- host ----------------
extern "C" void kernel_launch(void* const* d_in, const int* in_sizes, int n_in,
                              void* d_out, int out_size, void* d_ws, size_t ws_size,
                              hipStream_t stream) {
    const int*   edge_src  = (const int*)d_in[0];
    const int*   edge_dst  = (const int*)d_in[1];
    const float* edge_vec  = (const float*)d_in[2];
    const int*   atom_type = (const int*)d_in[3];
    const int*   node_gidx = (const int*)d_in[4];
    const float* embedding = (const float*)d_in[5];
    const float* phi_w1    = (const float*)d_in[6];
    const float* phi_b1    = (const float*)d_in[7];
    const float* phi_w2    = (const float*)d_in[8];
    const float* phi_b2    = (const float*)d_in[9];
    const float* filt_w    = (const float*)d_in[10];
    const float* filt_b    = (const float*)d_in[11];
    const float* upd_w1    = (const float*)d_in[12];
    const float* upd_b1    = (const float*)d_in[13];
    const float* upd_w2    = (const float*)d_in[14];
    const float* upd_b2    = (const float*)d_in[15];
    const float* U_w       = (const float*)d_in[16];
    const float* V_w       = (const float*)d_in[17];
    const float* out_w1    = (const float*)d_in[18];
    const float* out_b1    = (const float*)d_in[19];
    const float* out_w2    = (const float*)d_in[20];
    const float* out_b2    = (const float*)d_in[21];

    char* ws = (char*)d_ws;
    auto carve = [&](size_t bytes) -> char* {
        char* p = ws;
        ws += (bytes + 255) & ~(size_t)255;
        return p;
    };
    float*    s_buf = (float*)carve((size_t)N_NODES * DDIM * 4);
    float*    v_buf = (float*)carve((size_t)N_NODES * 384 * 4);
    float*    ds    = (float*)carve((size_t)N_NODES * DDIM * 4);
    float*    dv    = (float*)carve((size_t)N_NODES * 384 * 4);
    float*    gbuf  = (float*)carve((size_t)N_GRAPHS * DDIM * 4);
    unsigned* Bp1   = (unsigned*)carve(4 * 8 * 256 * 4);
    unsigned* Bp2   = (unsigned*)carve(4 * 24 * 256 * 4);
    unsigned* BpU   = (unsigned*)carve(4 * 8 * 256 * 4);
    unsigned* BpV   = (unsigned*)carve(4 * 8 * 256 * 4);
    unsigned* BpW1  = (unsigned*)carve(8 * 8 * 256 * 4);
    unsigned* BpW2  = (unsigned*)carve(4 * 24 * 256 * 4);

    // pack weights into WMMA B layout (bf16 pairs)
    k_pack_weights<<<(4 * 8 * 256 + 255) / 256, 256, 0, stream>>>(phi_w1, Bp1, 128, 128);
    k_pack_weights<<<(4 * 24 * 256 + 255) / 256, 256, 0, stream>>>(phi_w2, Bp2, 128, 384);
    k_pack_weights<<<(4 * 8 * 256 + 255) / 256, 256, 0, stream>>>(U_w, BpU, 128, 128);
    k_pack_weights<<<(4 * 8 * 256 + 255) / 256, 256, 0, stream>>>(V_w, BpV, 128, 128);
    k_pack_weights<<<(8 * 8 * 256 + 255) / 256, 256, 0, stream>>>(upd_w1, BpW1, 256, 128);
    k_pack_weights<<<(4 * 24 * 256 + 255) / 256, 256, 0, stream>>>(upd_w2, BpW2, 128, 384);

    // init s from embedding, v = 0
    long nS = (long)N_NODES * DDIM, nV = (long)N_NODES * 384;
    k_init_s<<<(int)((nS + 255) / 256), 256, 0, stream>>>(atom_type, embedding, s_buf);
    k_zero<<<(int)((nV + 255) / 256), 256, 0, stream>>>(v_buf, nV);

    for (int round = 0; round < 2; ++round) {
        k_zero<<<(int)((nS + 255) / 256), 256, 0, stream>>>(ds, nS);
        k_zero<<<(int)((nV + 255) / 256), 256, 0, stream>>>(dv, nV);
        k_edge<<<N_EDGES / 16, 32, EDGE_SMEM_BYTES, stream>>>(
            edge_src, edge_dst, edge_vec, s_buf, v_buf, ds, dv,
            Bp1, phi_b1, Bp2, phi_b2, filt_w, filt_b);
        k_node<<<N_NODES / 16, 32, NODE_SMEM_BYTES, stream>>>(
            s_buf, v_buf, ds, dv, BpU, BpV, BpW1, upd_b1, BpW2, upd_b2);
    }

    long nG = (long)N_GRAPHS * DDIM;
    k_zero<<<(int)((nG + 255) / 256), 256, 0, stream>>>(gbuf, nG);
    k_graph_accum<<<(int)((nS + 255) / 256), 256, 0, stream>>>(s_buf, node_gidx, gbuf);
    k_readout<<<N_GRAPHS, DDIM, 0, stream>>>(gbuf, out_w1, out_b1, out_w2, out_b2, (float*)d_out);
}
